// MultiHeadAttention_84318797955257
// MI455X (gfx1250) — matrix operand order repaired and checked by reading, not compiled, not run on hardware
//
#include <hip/hip_runtime.h>

// ---------------------------------------------------------------------------
// MI455X (gfx1250) multi-head attention with inverted (H*T)-axis softmax.
// bf16 WMMA everywhere; f32 accumulation; scores materialized in bf16.
// Tile movement uses gfx1250 async global->LDS copies with double buffering.
// ---------------------------------------------------------------------------

typedef __attribute__((ext_vector_type(16))) __bf16 v16bf;
typedef __attribute__((ext_vector_type(8)))  float  v8f;

constexpr int kBatch = 4, kT = 1024, kS = 1024, kD = 1024, kH = 16, kDH = 64;
constexpr int kBT = kBatch * kT;   // 4096 rows for flattened projections
constexpr int kHT = kH * kT;       // 16384 (softmax axis length)
constexpr int kCols = kBatch * kS; // 4096 softmax columns
constexpr int kChunks = 16;        // partial-reduction chunks along (h,t)
constexpr int kRowsPerChunk = kHT / kChunks;
constexpr float kNEG = -1e9f;
constexpr float kEPS = 1e-5f;

constexpr int BM = 128, BN = 128, BK = 32;
constexpr int LTS = 40;            // LDS tile row stride (ushort), 80B: 16B-aligned, depads banks

__device__ __forceinline__ unsigned short f2bf(float x) {
  unsigned u = __float_as_uint(x);
  u += 0x7FFFu + ((u >> 16) & 1u);            // round-to-nearest-even
  return (unsigned short)(u >> 16);
}
__device__ __forceinline__ float bf2f(unsigned short h) {
  return __uint_as_float(((unsigned)h) << 16);
}

// gfx1250 async global->LDS 16B copy (per active lane), tracked by ASYNCcnt.
__device__ __forceinline__ void async_copy16(const unsigned short* gsrc, unsigned short* ldst) {
  unsigned lds_addr = (unsigned)(uintptr_t)ldst;            // low 32 bits = LDS offset
  unsigned long long ga = (unsigned long long)(uintptr_t)gsrc;
  asm volatile("global_load_async_to_lds_b128 %0, %1, off"
               :: "v"(lds_addr), "v"(ga)
               : "memory");
}
__device__ __forceinline__ void wait_async0() {
  asm volatile("s_wait_asynccnt 0x0" ::: "memory");
}

union Frag { v16bf v; uint4 q[2]; };

// A fragment (16x32, MxK): lane<16 -> M=lane, K chunks {0-7,16-23};
// lane>=16 -> M=lane-16, K chunks {8-15,24-31}.  (ISA 7.12.2)
__device__ __forceinline__ void load_a_frag(const unsigned short* s, int m0, int lane, Frag& f) {
  int m = m0 + (lane & 15);
  int k0 = (lane >> 4) * 8;
  f.q[0] = *(const uint4*)&s[m * LTS + k0];
  f.q[1] = *(const uint4*)&s[m * LTS + k0 + 16];
}
// B fragment (32x16, KxN) from an N-major (K-contiguous) LDS tile:
// lane<16 -> N=lane, K chunks {0-7,16-23}; lane>=16 -> N=lane-16, K chunks {8-15,24-31}.
__device__ __forceinline__ void load_b_frag(const unsigned short* s, int n0, int lane, Frag& f) {
  int n = n0 + (lane & 15);
  int k0 = (lane >> 4) * 8;
  f.q[0] = *(const uint4*)&s[n * LTS + k0];
  f.q[1] = *(const uint4*)&s[n * LTS + k0 + 16];
}

// ---------------------------------------------------------------------------
// Generic NT GEMM: C[M,N] = alpha * A[M,K] * B[N,K]^T  (bf16 in, f32 acc)
// Block 128x128, 8 waves, each wave 32x64 (2x4 WMMA tiles).
// Async double-buffered LDS staging.
// ---------------------------------------------------------------------------
template <bool F32OUT>
__global__ void __launch_bounds__(256) gemm_nt_kernel(
    const unsigned short* __restrict__ A, const unsigned short* __restrict__ Bm,
    float alpha, int M, int N, int K, int lda, int ldb,
    unsigned short* __restrict__ Ch, float* __restrict__ Cf, int ldc)
{
  __shared__ __align__(16) unsigned short sA[2][BM * LTS];
  __shared__ __align__(16) unsigned short sB[2][BN * LTS];
  const int tid = threadIdx.x, lane = tid & 31, wid = tid >> 5;
  const int row0 = blockIdx.y * BM, col0 = blockIdx.x * BN;
  const int wrow = (wid >> 1) * 32, wcol = (wid & 1) * 64;

  // This thread's two 16B chunks of each 128x32 tile (512 chunks total).
  const int r0 = tid >> 2, cc0 = (tid & 3) * 8;
  const int r1 = (tid + 256) >> 2, cc1 = ((tid + 256) & 3) * 8;

  auto issue_tiles = [&](int buf, int kk) {
    async_copy16(&A[(size_t)(row0 + r0) * lda + kk + cc0], &sA[buf][r0 * LTS + cc0]);
    async_copy16(&A[(size_t)(row0 + r1) * lda + kk + cc1], &sA[buf][r1 * LTS + cc1]);
    async_copy16(&Bm[(size_t)(col0 + r0) * ldb + kk + cc0], &sB[buf][r0 * LTS + cc0]);
    async_copy16(&Bm[(size_t)(col0 + r1) * ldb + kk + cc1], &sB[buf][r1 * LTS + cc1]);
  };

  v8f acc[2][4] = {};

  issue_tiles(0, 0);
  wait_async0();
  __syncthreads();

  int p = 0;
  for (int kk = 0; kk < K; kk += BK, p ^= 1) {
    if (kk + BK < K) issue_tiles(p ^ 1, kk + BK);    // overlap fetch with WMMA
    Frag fa[2], fb[4];
#pragma unroll
    for (int ti = 0; ti < 2; ++ti) load_a_frag(sA[p], wrow + ti * 16, lane, fa[ti]);
#pragma unroll
    for (int tj = 0; tj < 4; ++tj) load_b_frag(sB[p], wcol + tj * 16, lane, fb[tj]);
#pragma unroll
    for (int ti = 0; ti < 2; ++ti)
#pragma unroll
      for (int tj = 0; tj < 4; ++tj)
        acc[ti][tj] = __builtin_amdgcn_wmma_f32_16x16x32_bf16(
            false, fa[ti].v, false, fb[tj].v, (short)0, acc[ti][tj], false, false);
    wait_async0();
    __syncthreads();
  }
#pragma unroll
  for (int ti = 0; ti < 2; ++ti)
#pragma unroll
    for (int tj = 0; tj < 4; ++tj)
#pragma unroll
      for (int r = 0; r < 8; ++r) {
        int row = row0 + wrow + ti * 16 + ((lane < 16) ? r : r + 8);
        int col = col0 + wcol + tj * 16 + (lane & 15);
        float val = acc[ti][tj][r] * alpha;
        if (F32OUT) Cf[(size_t)row * ldc + col] = val;
        else        Ch[(size_t)row * ldc + col] = f2bf(val);
      }
}

// ---------------------------------------------------------------------------
// Scores: per (b,h)   S[t,s] = Qh[t,:]·Kh[s,:] + bias[h,t,s], mask -> NEG
// ---------------------------------------------------------------------------
__global__ void __launch_bounds__(256) scores_kernel(
    const unsigned short* __restrict__ qh, const unsigned short* __restrict__ kh,
    const float* __restrict__ bias, const unsigned char* __restrict__ mask,
    unsigned short* __restrict__ scores)
{
  __shared__ __align__(16) unsigned short sA[2][BM * LTS];
  __shared__ __align__(16) unsigned short sB[2][BN * LTS];
  const int tid = threadIdx.x, lane = tid & 31, wid = tid >> 5;
  const int bh = blockIdx.z, b = bh >> 4, h = bh & 15;
  const int t0 = blockIdx.y * BM, s0 = blockIdx.x * BN;
  const unsigned short* A  = qh + (size_t)b * kT * kD + h * kDH;
  const unsigned short* Bm = kh + (size_t)b * kS * kD + h * kDH;
  const int wrow = (wid >> 1) * 32, wcol = (wid & 1) * 64;

  const int r0 = tid >> 2, cc0 = (tid & 3) * 8;
  const int r1 = (tid + 256) >> 2, cc1 = ((tid + 256) & 3) * 8;
  auto issue_tiles = [&](int buf, int kk) {
    async_copy16(&A[(size_t)(t0 + r0) * kD + kk + cc0], &sA[buf][r0 * LTS + cc0]);
    async_copy16(&A[(size_t)(t0 + r1) * kD + kk + cc1], &sA[buf][r1 * LTS + cc1]);
    async_copy16(&Bm[(size_t)(s0 + r0) * kD + kk + cc0], &sB[buf][r0 * LTS + cc0]);
    async_copy16(&Bm[(size_t)(s0 + r1) * kD + kk + cc1], &sB[buf][r1 * LTS + cc1]);
  };

  v8f acc[2][4] = {};
  issue_tiles(0, 0);
  wait_async0();
  __syncthreads();
  int p = 0;
  for (int kk = 0; kk < kDH; kk += BK, p ^= 1) {
    if (kk + BK < kDH) issue_tiles(p ^ 1, kk + BK);
    Frag fa[2], fb[4];
#pragma unroll
    for (int ti = 0; ti < 2; ++ti) load_a_frag(sA[p], wrow + ti * 16, lane, fa[ti]);
#pragma unroll
    for (int tj = 0; tj < 4; ++tj) load_b_frag(sB[p], wcol + tj * 16, lane, fb[tj]);
#pragma unroll
    for (int ti = 0; ti < 2; ++ti)
#pragma unroll
      for (int tj = 0; tj < 4; ++tj)
        acc[ti][tj] = __builtin_amdgcn_wmma_f32_16x16x32_bf16(
            false, fa[ti].v, false, fb[tj].v, (short)0, acc[ti][tj], false, false);
    wait_async0();
    __syncthreads();
  }
  unsigned short* Srow = scores + (size_t)bh * kT * kS;
#pragma unroll
  for (int ti = 0; ti < 2; ++ti)
#pragma unroll
    for (int tj = 0; tj < 4; ++tj)
#pragma unroll
      for (int r = 0; r < 8; ++r) {
        int t = t0 + wrow + ti * 16 + ((lane < 16) ? r : r + 8);
        int s = s0 + wcol + tj * 16 + (lane & 15);
        float x = acc[ti][tj][r] + bias[((size_t)h * kT + t) * kS + s];
        if (mask[(size_t)t * kS + s]) x = kNEG;
        Srow[(size_t)t * kS + s] = f2bf(x);
      }
}

// ---------------------------------------------------------------------------
// Column softmax stats, stage 1: partial (max, sum) over a 1024-row chunk of
// the (h,t) axis for each (b,s) column. 65536 threads -> bandwidth bound.
// ---------------------------------------------------------------------------
__global__ void __launch_bounds__(256) colstats_partial_kernel(
    const unsigned short* __restrict__ scores,
    float* __restrict__ pmax, float* __restrict__ psum)
{
  int idx = blockIdx.x * 256 + threadIdx.x;      // 0 .. kCols*kChunks-1
  int col = idx & (kCols - 1);                   // consecutive threads -> consecutive s
  int chunk = idx >> 12;
  int b = col >> 10, s = col & (kS - 1);
  const unsigned short* base =
      scores + (size_t)b * kHT * kS + (size_t)chunk * kRowsPerChunk * kS + s;
  float m = -3.4e38f, sum = 0.f;
  for (int r = 0; r < kRowsPerChunk; ++r) {
    float x = bf2f(base[(size_t)r * kS]);
    if (x > m) { sum = sum * __expf(m - x) + 1.f; m = x; }
    else       { sum += __expf(x - m); }
  }
  pmax[idx] = m;
  psum[idx] = sum;
}

// Stage 2: merge the 16 partials per column.
__global__ void __launch_bounds__(256) colstats_final_kernel(
    const float* __restrict__ pmax, const float* __restrict__ psum,
    float* __restrict__ colmax, float* __restrict__ colsum)
{
  int col = blockIdx.x * 256 + threadIdx.x;      // 0..kCols-1
  float m = -3.4e38f;
#pragma unroll
  for (int c = 0; c < kChunks; ++c) m = fmaxf(m, pmax[c * kCols + col]);
  float sum = 0.f;
#pragma unroll
  for (int c = 0; c < kChunks; ++c)
    sum += psum[c * kCols + col] * __expf(pmax[c * kCols + col] - m);
  colmax[col] = m;
  colsum[col] = sum;
}

// ---------------------------------------------------------------------------
// Row pass: a = exp(x - colmax)/colsum ; mask -> 0 ; a /= (row_sum + eps)
// One block per (b,h,t) row; in-place bf16 overwrite of the score row.
// ---------------------------------------------------------------------------
__global__ void __launch_bounds__(256) rownorm_kernel(
    unsigned short* __restrict__ scores, const float* __restrict__ colmax,
    const float* __restrict__ colsum, const unsigned char* __restrict__ mask)
{
  const int row = blockIdx.x;                 // (b*H + h)*T + t
  const int b = row >> 14;                    // / (H*T)
  const int t = row & (kT - 1);
  unsigned short* rp = scores + (size_t)row * kS;
  const float* cm = colmax + (size_t)b * kS;
  const float* cs = colsum + (size_t)b * kS;
  const int tid = threadIdx.x, lane = tid & 31, wid = tid >> 5;

  float p[4]; float lsum = 0.f;
#pragma unroll
  for (int j = 0; j < 4; ++j) {
    int s = tid + j * 256;
    float x = bf2f(rp[s]);
    float e = __expf(x - cm[s]) / cs[s];
    if (mask[(size_t)t * kS + s]) e = 0.f;
    p[j] = e; lsum += e;
  }
#pragma unroll
  for (int off = 16; off > 0; off >>= 1) lsum += __shfl_xor(lsum, off, 32);
  __shared__ float wsum[8];
  if (lane == 0) wsum[wid] = lsum;
  __syncthreads();
  float tot = 0.f;
#pragma unroll
  for (int w = 0; w < 8; ++w) tot += wsum[w];
  float scale = 1.f / (tot + kEPS);
#pragma unroll
  for (int j = 0; j < 4; ++j) rp[tid + j * 256] = f2bf(p[j] * scale);
}

// ---------------------------------------------------------------------------
// PV: per (b,h)  O[t,dh] = sum_s a[t,s] * Vh[s,dh].
// A tile async double-buffered; V tile transposed through LDS (register path).
// Block 128x64, 8 waves of 32x32 (2x2 tiles).
// ---------------------------------------------------------------------------
constexpr int PBM = 128, PBK = 32;
__global__ void __launch_bounds__(256) pv_kernel(
    const unsigned short* __restrict__ probs, const unsigned short* __restrict__ vh,
    unsigned short* __restrict__ obf)
{
  __shared__ __align__(16) unsigned short sA[2][PBM * LTS];
  __shared__ __align__(16) unsigned short sB[2][kDH * LTS];
  const int tid = threadIdx.x, lane = tid & 31, wid = tid >> 5;
  const int bh = blockIdx.z, b = bh >> 4, h = bh & 15;
  const int t0 = blockIdx.x * PBM;
  const unsigned short* A = probs + (size_t)bh * kT * kS;            // [T][S]
  const unsigned short* V = vh + (size_t)b * kS * kD + h * kDH;      // [S][64], stride kD
  const int wrow = (wid >> 1) * 32, wcol = (wid & 1) * 32;

  const int rA0 = tid >> 2, ccA0 = (tid & 3) * 8;
  const int rA1 = (tid + 256) >> 2, ccA1 = ((tid + 256) & 3) * 8;
  const int sV = tid >> 3, dcV = (tid & 7) * 8;

  auto issueA = [&](int buf, int s0) {
    async_copy16(&A[(size_t)(t0 + rA0) * kS + s0 + ccA0], &sA[buf][rA0 * LTS + ccA0]);
    async_copy16(&A[(size_t)(t0 + rA1) * kS + s0 + ccA1], &sA[buf][rA1 * LTS + ccA1]);
  };
  auto loadB = [&](int buf, int s0) {                 // 32(s) x 64(dh) -> sB[dh][s]
    uint4 x = *(const uint4*)&V[(size_t)(s0 + sV) * kD + dcV];
    const unsigned short* u = (const unsigned short*)&x;
#pragma unroll
    for (int j = 0; j < 8; ++j) sB[buf][(dcV + j) * LTS + sV] = u[j];
  };

  v8f acc[2][2] = {};
  issueA(0, 0); loadB(0, 0);
  wait_async0();
  __syncthreads();
  int p = 0;
  for (int s0 = 0; s0 < kS; s0 += PBK, p ^= 1) {
    if (s0 + PBK < kS) { issueA(p ^ 1, s0 + PBK); loadB(p ^ 1, s0 + PBK); }
    Frag fa[2], fb[2];
#pragma unroll
    for (int ti = 0; ti < 2; ++ti) load_a_frag(sA[p], wrow + ti * 16, lane, fa[ti]);
#pragma unroll
    for (int tj = 0; tj < 2; ++tj) load_b_frag(sB[p], wcol + tj * 16, lane, fb[tj]);
#pragma unroll
    for (int ti = 0; ti < 2; ++ti)
#pragma unroll
      for (int tj = 0; tj < 2; ++tj)
        acc[ti][tj] = __builtin_amdgcn_wmma_f32_16x16x32_bf16(
            false, fa[ti].v, false, fb[tj].v, (short)0, acc[ti][tj], false, false);
    wait_async0();
    __syncthreads();
  }
#pragma unroll
  for (int ti = 0; ti < 2; ++ti)
#pragma unroll
    for (int tj = 0; tj < 2; ++tj)
#pragma unroll
      for (int r = 0; r < 8; ++r) {
        int t = t0 + wrow + ti * 16 + ((lane < 16) ? r : r + 8);
        int col = wcol + tj * 16 + (lane & 15);
        obf[(size_t)(b * kT + t) * kD + h * kDH + col] = f2bf(acc[ti][tj][r]);
      }
}

// ---------------------------------------------------------------------------
// f32 -> bf16 conversion (vectorized, n multiple of 4)
// ---------------------------------------------------------------------------
__global__ void __launch_bounds__(256) cvt_bf16_kernel(
    const float* __restrict__ in, unsigned short* __restrict__ out, int n)
{
  int idx = (blockIdx.x * 256 + threadIdx.x) * 4;
  if (idx >= n) return;
  float4 v = *(const float4*)&in[idx];
  union { unsigned short r[4]; uint2 u; } o;
  o.r[0] = f2bf(v.x); o.r[1] = f2bf(v.y); o.r[2] = f2bf(v.z); o.r[3] = f2bf(v.w);
  *(uint2*)&out[idx] = o.u;
}

// ---------------------------------------------------------------------------
extern "C" void kernel_launch(void* const* d_in, const int* in_sizes, int n_in,
                              void* d_out, int out_size, void* d_ws, size_t ws_size,
                              hipStream_t stream)
{
  const float* q  = (const float*)d_in[0];
  const float* k  = (const float*)d_in[1];
  const float* v  = (const float*)d_in[2];
  const unsigned char* mask = (const unsigned char*)d_in[3];
  const float* bias = (const float*)d_in[4];
  const float* Wq = (const float*)d_in[5];
  const float* Wk = (const float*)d_in[6];
  const float* Wv = (const float*)d_in[7];
  const float* Wo = (const float*)d_in[8];
  float* out = (float*)d_out;

  char* p = (char*)d_ws;
  auto takeH = [&](size_t elems) {                       // bf16 buffer
    unsigned short* r = (unsigned short*)p;
    p += (elems * 2 + 255) & ~(size_t)255;
    return r;
  };
  auto takeF = [&](size_t elems) {                       // f32 buffer
    float* r = (float*)p;
    p += (elems * 4 + 255) & ~(size_t)255;
    return r;
  };
  const size_t nQ = (size_t)kBT * kD;    // 4.19M elems
  const size_t nW = (size_t)kD * kD;     // 1.05M elems
  unsigned short* qbf = takeH(nQ);
  unsigned short* kbf = takeH(nQ);
  unsigned short* vbf = takeH(nQ);
  unsigned short* qh  = takeH(nQ);
  unsigned short* kh  = takeH(nQ);
  unsigned short* vhB = takeH(nQ);
  unsigned short* obf = takeH(nQ);
  unsigned short* wqb = takeH(nW);
  unsigned short* wkb = takeH(nW);
  unsigned short* wvb = takeH(nW);
  unsigned short* wob = takeH(nW);
  float* colmax = takeF((size_t)kCols);
  float* colsum = takeF((size_t)kCols);
  float* pmax   = takeF((size_t)kCols * kChunks);
  float* psum   = takeF((size_t)kCols * kChunks);
  unsigned short* scores = takeH((size_t)kBatch * kH * kT * kS);   // 128 MB

  // 1) convert everything to bf16
  cvt_bf16_kernel<<<(int)(nQ / 1024), 256, 0, stream>>>(q, qbf, (int)nQ);
  cvt_bf16_kernel<<<(int)(nQ / 1024), 256, 0, stream>>>(k, kbf, (int)nQ);
  cvt_bf16_kernel<<<(int)(nQ / 1024), 256, 0, stream>>>(v, vbf, (int)nQ);
  cvt_bf16_kernel<<<(int)(nW / 1024), 256, 0, stream>>>(Wq, wqb, (int)nW);
  cvt_bf16_kernel<<<(int)(nW / 1024), 256, 0, stream>>>(Wk, wkb, (int)nW);
  cvt_bf16_kernel<<<(int)(nW / 1024), 256, 0, stream>>>(Wv, wvb, (int)nW);
  cvt_bf16_kernel<<<(int)(nW / 1024), 256, 0, stream>>>(Wo, wob, (int)nW);

  // 2) projections (NT GEMM); DH^-0.5 = 0.125 folded into Q
  dim3 gp(kD / BN, kBT / BM);
  gemm_nt_kernel<false><<<gp, 256, 0, stream>>>(qbf, wqb, 0.125f, kBT, kD, kD, kD, kD, qh,  nullptr, kD);
  gemm_nt_kernel<false><<<gp, 256, 0, stream>>>(kbf, wkb, 1.0f,   kBT, kD, kD, kD, kD, kh,  nullptr, kD);
  gemm_nt_kernel<false><<<gp, 256, 0, stream>>>(vbf, wvb, 1.0f,   kBT, kD, kD, kD, kD, vhB, nullptr, kD);

  // 3) scores + bias + mask  -> bf16 [B,H,T,S]
  scores_kernel<<<dim3(kS / BN, kT / BM, kBatch * kH), 256, 0, stream>>>(qh, kh, bias, mask, scores);

  // 4) column softmax stats over (h,t) per (b,s): partial then merge
  colstats_partial_kernel<<<(kCols * kChunks) / 256, 256, 0, stream>>>(scores, pmax, psum);
  colstats_final_kernel<<<kCols / 256, 256, 0, stream>>>(pmax, psum, colmax, colsum);

  // 5) normalize columns, mask-zero, renormalize rows (in-place bf16 probs)
  rownorm_kernel<<<kBatch * kH * kT, 256, 0, stream>>>(scores, colmax, colsum, mask);

  // 6) PV -> attention output [B*T, D] bf16
  pv_kernel<<<dim3(kT / PBM, 1, kBatch * kH), 256, 0, stream>>>(scores, vhB, obf);

  // 7) output projection -> f32 d_out
  gemm_nt_kernel<true><<<gp, 256, 0, stream>>>(obf, wob, 1.0f, kBT, kD, kD, kD, kD, nullptr, out, kD);
}
